// CausalSelfAttention_68255620268584
// MI455X (gfx1250) — matrix operand order repaired and checked by reading, not compiled, not run on hardware
//
#include <hip/hip_runtime.h>
#include <hip/hip_bf16.h>

typedef unsigned short u16;
typedef __attribute__((ext_vector_type(16))) __bf16 v16bf;
typedef __attribute__((ext_vector_type(8)))  float  v8f;
typedef __attribute__((ext_vector_type(4)))  int    v4i;

#if __has_builtin(__builtin_amdgcn_global_load_async_to_lds_b128)
#define HAVE_ASYNC_LDS 1
#else
#define HAVE_ASYNC_LDS 0
#endif

#define AS1 __attribute__((address_space(1)))
#define AS3 __attribute__((address_space(3)))

union FragAB {
  v16bf bf;
  uint4 q[2];
  u16   u[16];
};
union FragC {
  v8f   v;
  float f[8];
};
union V4U16 {
  uint2 q;
  u16   u[4];
};
union V8U16 {
  uint4 q;
  u16   u[8];
};

__device__ __forceinline__ u16 f2bf(float f) {
  union { float f; unsigned int u; } c; c.f = f;
  unsigned int u = c.u;
  u += 0x7FFFu + ((u >> 16) & 1u);   // round to nearest even
  return (u16)(u >> 16);
}

// ---- 16-lane (half-wave) butterfly reductions via DPP16 (pure VALU) --------
template <int CTRL>
__device__ __forceinline__ float dppmov(float x) {
  return __int_as_float(__builtin_amdgcn_update_dpp(
      0, __float_as_int(x), CTRL, 0xF, 0xF, true));
}
__device__ __forceinline__ float red16max(float x) {
  x = fmaxf(x, dppmov<0xB1>(x));    // quad_perm [1,0,3,2]
  x = fmaxf(x, dppmov<0x4E>(x));    // quad_perm [2,3,0,1]
  x = fmaxf(x, dppmov<0x141>(x));   // row_half_mirror
  x = fmaxf(x, dppmov<0x140>(x));   // row_mirror
  return x;
}
__device__ __forceinline__ float red16sum(float x) {
  x += dppmov<0xB1>(x);
  x += dppmov<0x4E>(x);
  x += dppmov<0x141>(x);
  x += dppmov<0x140>(x);
  return x;
}

__device__ __forceinline__ void wait_async0() {
#if HAVE_ASYNC_LDS
#if __has_builtin(__builtin_amdgcn_s_wait_asynccnt)
  __builtin_amdgcn_s_wait_asynccnt(0);
#else
  asm volatile("s_wait_asynccnt 0" ::: "memory");
#endif
#endif
}

#if HAVE_ASYNC_LDS
__device__ __forceinline__ void async_cp16(const u16* g, u16* l) {
  __builtin_amdgcn_global_load_async_to_lds_b128(
      (AS1 v4i*)g, (AS3 v4i*)l, 0, 0);
}
#endif

// ---------------------------------------------------------------------------
// casts
// ---------------------------------------------------------------------------
__global__ __launch_bounds__(256) void cast_bf16_k(const float* __restrict__ in,
                                                   u16* __restrict__ out, int n) {
  int i = blockIdx.x * 256 + threadIdx.x;
  if (i < n) out[i] = f2bf(in[i]);
}

// out[n][k] = bf16(in[k][n]),  1024 x 1024
__global__ __launch_bounds__(256) void transpose_cast_k(const float* __restrict__ in,
                                                        u16* __restrict__ out) {
  int i = blockIdx.x * 256 + threadIdx.x;   // 0..1M-1
  int k = i >> 10;
  int n = i & 1023;
  out[((size_t)n << 10) + k] = f2bf(in[((size_t)k << 10) + n]);
}

// ---------------------------------------------------------------------------
// GEMM: Out[m][n] = sum_k A[m][k] * Bt[n][k]    (A,Bt bf16)
// M=8192, N=1024, K=1024.  Block tile 128x128, 8 waves (4x2), wave tile 32x64.
// Double-buffered LDS; async global->LDS staging when available.
// MODE 0: store bf16 head-major [B,NH,T,64];  MODE 1: store fp32 [M][N].
// ---------------------------------------------------------------------------
template <int MODE>
__global__ __launch_bounds__(256) void gemm_k(const u16* __restrict__ A,
                                              const u16* __restrict__ Bt,
                                              void* __restrict__ Out) {
  const int K = 1024, N = 1024;
  __shared__ __align__(16) u16 sA[2][128 * 40];
  __shared__ __align__(16) u16 sB[2][128 * 40];

  const int m0   = blockIdx.x * 128;
  const int n0   = blockIdx.y * 128;
  const int tid  = threadIdx.x;
  const int lane = tid & 31;
  const int w    = tid >> 5;
  const int wm   = w & 3;
  const int wn   = w >> 2;
  const int hi   = lane >> 4;
  const int l16  = lane & 15;
  const int srow = tid >> 1;    // staging row 0..127
  const int shalf = tid & 1;

  const u16* gA = A  + (size_t)(m0 + srow) * K + shalf * 16;
  const u16* gB = Bt + (size_t)(n0 + srow) * K + shalf * 16;
  const int sOff = srow * 40 + shalf * 16;

  FragC acc[2][4];
#pragma unroll
  for (int i = 0; i < 2; i++)
#pragma unroll
    for (int j = 0; j < 4; j++)
#pragma unroll
      for (int r = 0; r < 8; r++) acc[i][j].f[r] = 0.f;

#if HAVE_ASYNC_LDS
  async_cp16(gA,     &sA[0][sOff]);
  async_cp16(gA + 8, &sA[0][sOff + 8]);
  async_cp16(gB,     &sB[0][sOff]);
  async_cp16(gB + 8, &sB[0][sOff + 8]);
#else
  {
    const uint4* srcA = (const uint4*)gA;
    uint4* dstA = (uint4*)(&sA[0][sOff]);
    dstA[0] = srcA[0];
    dstA[1] = srcA[1];
    const uint4* srcB = (const uint4*)gB;
    uint4* dstB = (uint4*)(&sB[0][sOff]);
    dstB[0] = srcB[0];
    dstB[1] = srcB[1];
  }
#endif

  for (int it = 0; it < 32; ++it) {
    const int cur = it & 1;
    wait_async0();            // staged tile for 'cur' has landed (async path)
    __syncthreads();          // visible to all waves; prev reads of 'cur^1' done

#if HAVE_ASYNC_LDS
    if (it + 1 < 32) {
      const u16* nA = gA + (it + 1) * 32;
      const u16* nB = gB + (it + 1) * 32;
      const int nb = cur ^ 1;
      async_cp16(nA,     &sA[nb][sOff]);
      async_cp16(nA + 8, &sA[nb][sOff + 8]);
      async_cp16(nB,     &sB[nb][sOff]);
      async_cp16(nB + 8, &sB[nb][sOff + 8]);
    }
#else
    uint4 ra0, ra1, rb0, rb1;
    if (it + 1 < 32) {
      const uint4* srcA = (const uint4*)(gA + (it + 1) * 32);
      const uint4* srcB = (const uint4*)(gB + (it + 1) * 32);
      ra0 = srcA[0]; ra1 = srcA[1];
      rb0 = srcB[0]; rb1 = srcB[1];
    }
#endif
    if (it + 2 < 32) {
      __builtin_prefetch(gA + (it + 2) * 32, 0, 3);
      __builtin_prefetch(gB + (it + 2) * 32, 0, 3);
    }

    FragAB af[2], bfr[4];
#pragma unroll
    for (int i = 0; i < 2; i++) {
      int r = wm * 32 + i * 16 + l16;
      af[i].q[0] = *(const uint4*)(&sA[cur][r * 40 + hi * 8]);
      af[i].q[1] = *(const uint4*)(&sA[cur][r * 40 + 16 + hi * 8]);
    }
#pragma unroll
    for (int j = 0; j < 4; j++) {
      int c = wn * 64 + j * 16 + l16;
      const uint4* p = (const uint4*)(&sB[cur][c * 40 + hi * 8]);
      bfr[j].q[0] = p[0];
      bfr[j].q[1] = p[2];
    }
#pragma unroll
    for (int i = 0; i < 2; i++)
#pragma unroll
      for (int j = 0; j < 4; j++)
        acc[i][j].v = __builtin_amdgcn_wmma_f32_16x16x32_bf16(
            false, af[i].bf, false, bfr[j].bf, (short)0, acc[i][j].v, false, false);

#if !HAVE_ASYNC_LDS
    if (it + 1 < 32) {
      const int nb = cur ^ 1;
      uint4* dstA = (uint4*)(&sA[nb][sOff]);
      dstA[0] = ra0; dstA[1] = ra1;
      uint4* dstB = (uint4*)(&sB[nb][sOff]);
      dstB[0] = rb0; dstB[1] = rb1;
    }
#endif
  }

#pragma unroll
  for (int i = 0; i < 2; i++) {
    int grow_base = m0 + wm * 32 + i * 16 + hi * 8;
#pragma unroll
    for (int j = 0; j < 4; j++) {
      int gcol = n0 + wn * 64 + j * 16 + l16;
#pragma unroll
      for (int r = 0; r < 8; r++) {
        int grow = grow_base + r;
        if (MODE == 0) {
          u16* O = (u16*)Out;
          int b = grow >> 11;
          int t = grow & 2047;
          int h = gcol >> 6;
          int d = gcol & 63;
          O[(((size_t)(b * 16 + h) * 2048) + t) * 64 + d] = f2bf(acc[i][j].f[r]);
        } else {
          float* O = (float*)Out;
          O[(size_t)grow * N + gcol] = acc[i][j].f[r];
        }
      }
    }
  }
}

// ---------------------------------------------------------------------------
// Flash attention: grid (T/128, NH, B), 256 thr. Wave w owns 16 q-rows.
// 64-key chunks: sK [64 keys][72], sVt [64 d][72], per-wave sP [16][72].
// ---------------------------------------------------------------------------
__global__ __launch_bounds__(256) void attn_k(const u16* __restrict__ Q,
                                              const u16* __restrict__ Kh,
                                              const u16* __restrict__ Vh,
                                              u16* __restrict__ Yb) {
  const int T = 2048, HS = 64, NH = 16, C = 1024;
  __shared__ __align__(16) u16 sK[64 * 72];
  __shared__ __align__(16) u16 sVt[64 * 72];
  __shared__ __align__(16) u16 sP[8 * 16 * 72];

  const int qb   = blockIdx.x;
  const int h    = blockIdx.y;
  const int b    = blockIdx.z;
  const int tid  = threadIdx.x;
  const int lane = tid & 31;
  const int w    = tid >> 5;
  const int hi   = lane >> 4;
  const int l16  = lane & 15;

  const size_t base = (size_t)(b * NH + h) * T * HS;
  const int row0 = qb * 128 + w * 16;

  // staging indices
  const int krow = tid >> 2;   // 0..63 (K rows), 16 feats each
  const int kseg = tid & 3;
  const int kp   = tid >> 3;   // 0..31 (V key pairs)
  const int dsg  = tid & 7;    // d = dsg*8..+8

  FragAB aq[2];
  {
    const u16* qptr = Q + base + (size_t)(row0 + l16) * 64;
#pragma unroll
    for (int j = 0; j < 2; j++) {
      aq[j].q[0] = *(const uint4*)(qptr + j * 32 + hi * 8);
      aq[j].q[1] = *(const uint4*)(qptr + j * 32 + 16 + hi * 8);
    }
  }

  float mrun[8], lrun[8];
  FragC o[4];
#pragma unroll
  for (int r = 0; r < 8; r++) { mrun[r] = -1e30f; lrun[r] = 0.f; }
#pragma unroll
  for (int dt = 0; dt < 4; dt++)
#pragma unroll
    for (int r = 0; r < 8; r++) o[dt].f[r] = 0.f;

  const int nchunks = (qb + 1) * 2;   // keys 0 .. (qb+1)*128 in chunks of 64
  for (int ch = 0; ch < nchunks; ch++) {
    const int key0 = ch * 64;
    __syncthreads();
    // stage K chunk: [64 keys][64 feats], 16 feats per thread
    {
      const uint4* src = (const uint4*)(Kh + base + (size_t)(key0 + krow) * 64 + kseg * 16);
      uint4* dst = (uint4*)(sK + krow * 72 + kseg * 16);
      dst[0] = src[0];
      dst[1] = src[1];
    }
    // stage V chunk transposed; pack two keys per b32 store
    {
      V8U16 a0, a1;
      a0.q = *(const uint4*)(Vh + base + (size_t)(key0 + 2 * kp) * 64 + dsg * 8);
      a1.q = *(const uint4*)(Vh + base + (size_t)(key0 + 2 * kp + 1) * 64 + dsg * 8);
#pragma unroll
      for (int j = 0; j < 8; j++) {
        *(unsigned int*)(sVt + (dsg * 8 + j) * 72 + 2 * kp) =
            (unsigned int)a0.u[j] | ((unsigned int)a1.u[j] << 16);
      }
    }
    __syncthreads();

    // S = Q @ K^T : 16 rows x 64 keys (four 16-wide tiles)
    FragC s[4];
#pragma unroll
    for (int kt = 0; kt < 4; kt++) {
#pragma unroll
      for (int r = 0; r < 8; r++) s[kt].f[r] = 0.f;
#pragma unroll
      for (int j = 0; j < 2; j++) {
        FragAB bk;
        const uint4* p = (const uint4*)(sK + (kt * 16 + l16) * 72 + j * 32 + hi * 8);
        bk.q[0] = p[0];
        bk.q[1] = p[2];
        s[kt].v = __builtin_amdgcn_wmma_f32_16x16x32_bf16(
            false, aq[j].bf, false, bk.bf, (short)0, s[kt].v, false, false);
      }
    }

    // masked online softmax; row reductions via DPP; store P as we go
    u16* sPw = sP + w * 16 * 72;
#pragma unroll
    for (int r = 0; r < 8; r++) {
      int grow = row0 + hi * 8 + r;
      float v[4];
      bool ok[4];
      float vm = -1e30f;
#pragma unroll
      for (int kt = 0; kt < 4; kt++) {
        ok[kt] = (key0 + kt * 16 + l16) <= grow;
        v[kt] = s[kt].f[r] * 0.125f;
        vm = fmaxf(vm, ok[kt] ? v[kt] : -1e30f);
      }
      vm = red16max(vm);
      float newm = fmaxf(mrun[r], vm);
      float alpha = __expf(mrun[r] - newm);
      float rs = 0.f;
      int mrow = hi * 8 + r;
#pragma unroll
      for (int kt = 0; kt < 4; kt++) {
        float e = ok[kt] ? __expf(v[kt] - newm) : 0.f;
        rs += e;
        sPw[mrow * 72 + kt * 16 + l16] = f2bf(e);
      }
      rs = red16sum(rs);
      lrun[r] = lrun[r] * alpha + rs;
      mrun[r] = newm;
#pragma unroll
      for (int dt = 0; dt < 4; dt++) o[dt].f[r] *= alpha;
    }
    asm volatile("s_wait_dscnt 0" ::: "memory");

    // P A-fragments: keys 0..31 and 32..63
    FragAB pf[2];
#pragma unroll
    for (int p = 0; p < 2; p++) {
      pf[p].q[0] = *(const uint4*)(sPw + l16 * 72 + p * 32 + hi * 8);
      pf[p].q[1] = *(const uint4*)(sPw + l16 * 72 + p * 32 + 16 + hi * 8);
    }

    // O += P @ V
#pragma unroll
    for (int dt = 0; dt < 4; dt++) {
#pragma unroll
      for (int p = 0; p < 2; p++) {
        FragAB bv;
        const uint4* q = (const uint4*)(sVt + (dt * 16 + l16) * 72 + p * 32 + hi * 8);
        bv.q[0] = q[0];
        bv.q[1] = q[2];
        o[dt].v = __builtin_amdgcn_wmma_f32_16x16x32_bf16(
            false, pf[p].bf, false, bv.bf, (short)0, o[dt].v, false, false);
      }
    }
  }

  // normalize and write Yb[b*T + row][h*64 + d]
#pragma unroll
  for (int r = 0; r < 8; r++) {
    float inv = 1.f / lrun[r];
    int grow = row0 + hi * 8 + r;
#pragma unroll
    for (int dt = 0; dt < 4; dt++) {
      int gcol = h * 64 + dt * 16 + l16;
      Yb[((size_t)(b * T + grow)) * C + gcol] = f2bf(o[dt].f[r] * inv);
    }
  }
}

// ---------------------------------------------------------------------------
// launch
// ---------------------------------------------------------------------------
extern "C" void kernel_launch(void* const* d_in, const int* in_sizes, int n_in,
                              void* d_out, int out_size, void* d_ws, size_t ws_size,
                              hipStream_t stream) {
  (void)in_sizes; (void)n_in; (void)out_size; (void)ws_size;
  const float* x  = (const float*)d_in[0];
  const float* Wk = (const float*)d_in[1];
  const float* Wq = (const float*)d_in[2];
  const float* Wv = (const float*)d_in[3];
  const float* Wp = (const float*)d_in[4];

  char* ws = (char*)d_ws;
  u16* xb  = (u16*)(ws);
  u16* WqT = (u16*)(ws + (16ull << 20));
  u16* WkT = (u16*)(ws + (18ull << 20));
  u16* WvT = (u16*)(ws + (20ull << 20));
  u16* WpT = (u16*)(ws + (22ull << 20));
  u16* Qb  = (u16*)(ws + (24ull << 20));
  u16* Kb  = (u16*)(ws + (40ull << 20));
  u16* Vb  = (u16*)(ws + (56ull << 20));
  u16* Yb  = (u16*)(ws + (72ull << 20));

  const int NX = 4 * 2048 * 1024;
  cast_bf16_k<<<NX / 256, 256, 0, stream>>>(x, xb, NX);
  transpose_cast_k<<<4096, 256, 0, stream>>>(Wq, WqT);
  transpose_cast_k<<<4096, 256, 0, stream>>>(Wk, WkT);
  transpose_cast_k<<<4096, 256, 0, stream>>>(Wv, WvT);
  transpose_cast_k<<<4096, 256, 0, stream>>>(Wp, WpT);

  dim3 gg(64, 8);
  gemm_k<0><<<gg, 256, 0, stream>>>(xb, WqT, Qb);
  gemm_k<0><<<gg, 256, 0, stream>>>(xb, WkT, Kb);
  gemm_k<0><<<gg, 256, 0, stream>>>(xb, WvT, Vb);

  dim3 ga(16, 16, 4);
  attn_k<<<ga, 256, 0, stream>>>(Qb, Kb, Vb, Yb);

  gemm_k<1><<<gg, 256, 0, stream>>>(Yb, WpT, d_out);
}
